// SpaceSimilarityRegularizedMultiDCNnew2_28166395527373
// MI455X (gfx1250) — matrix operand order repaired and checked by reading, not compiled, not run on hardware
//
#include <hip/hip_runtime.h>
#include <hip/hip_bf16.h>

// ---------------- problem constants ----------------
#define BB   8192
#define FF   16
#define VV   50000
#define DD   32
#define NEB  3
#define NLY  3
#define EOD  512
#define MM1  512
#define MM2  256
#define MM3  128
#define REGW 0.01f

typedef __attribute__((ext_vector_type(16))) __bf16 v16bf;
typedef __attribute__((ext_vector_type(8)))  float  v8f;

union FragB16 { unsigned int u[8]; v16bf v; };

__device__ __forceinline__ unsigned short f32_to_bf16u(float f) {
    unsigned int u = __float_as_uint(f);
    u += 0x7fffu + ((u >> 16) & 1u);           // round-to-nearest-even
    return (unsigned short)(u >> 16);
}
__device__ __forceinline__ float bf16u_to_f32(unsigned short s) {
    return __uint_as_float(((unsigned int)s) << 16);
}

// ---------------- weight transpose + bf16 convert ----------------
// src: batch of K x N row-major fp32; dst: batch of N x K row-major bf16
__global__ void transpose_to_bf16(const float* __restrict__ src,
                                  unsigned short* __restrict__ dst,
                                  int K, int N) {
    long long base = (long long)blockIdx.z * K * N;
    int gid = blockIdx.x * blockDim.x + threadIdx.x;
    if (gid >= K * N) return;
    int n = gid % N, k = gid / N;
    dst[base + (long long)n * K + k] = f32_to_bf16u(src[base + gid]);
}

// ---------------- embedding gather ----------------
__global__ void gather_emb(const int* __restrict__ x, const float* __restrict__ emb,
                           float* __restrict__ esF, unsigned short* __restrict__ esH) {
    long long gid = (long long)blockIdx.x * blockDim.x + threadIdx.x;
    const long long total = (long long)NEB * BB * FF * DD;
    if (gid >= total) return;
    int d = (int)(gid & (DD - 1));
    long long t = gid >> 5;            // /D
    int k = (int)(t & (FF - 1));
    t >>= 4;                           // /F
    int b = (int)(t & (BB - 1));
    int e = (int)(t >> 13);            // /B
    long long row = (long long)x[b * FF + k] + (long long)k * VV;
    float v = emb[((long long)e * FF * VV + row) * DD + d];
    esF[gid] = v;
    esH[gid] = f32_to_bf16u(v);
}

// ---------------- WMMA bf16 GEMM with fused epilogues ----------------
// C(M x N) = A(M x K, bf16 row-major) * W (stored transposed: N x K bf16 row-major)
// mode 0: out = x0*(C+bias)+xl        -> write fp32 + bf16   (DCN cross layer)
// mode 1: out = relu(C+bias)          -> write fp32 only     (per-branch mlp1 head)
// mode 2: out = relu(C+bias)          -> write bf16 only     (MLP tower)
__global__ __launch_bounds__(256)
void gemm_bf16_wmma(const unsigned short* __restrict__ A, long long aStride, int lda,
                    const unsigned short* __restrict__ WT, long long wStride,
                    const float* __restrict__ bias, long long biasStride,
                    const float* __restrict__ x0f, const float* __restrict__ xlf, long long xStride,
                    float* __restrict__ outF, unsigned short* __restrict__ outH, long long outStride,
                    int N, int K, int mode) {
    const int e = blockIdx.z;
    const unsigned short* Ae = A + (long long)e * aStride;
    const unsigned short* We = WT + (long long)e * wStride;
    const float* be = bias + (long long)e * biasStride;

    const int tid  = threadIdx.x;
    const int lane = tid & 31;
    const int wave = tid >> 5;
    const int wm = wave & 3;         // 4 waves along M
    const int wn = wave >> 2;        // 2 waves along N
    const int rowT = blockIdx.x * 128 + wm * 32;
    const int colT = blockIdx.y * 128 + wn * 64;
    const int r = lane & 15;         // row (A) / col (B,C) within 16
    const int h = lane >> 4;         // lane-half selector

    v8f zero = {0.f, 0.f, 0.f, 0.f, 0.f, 0.f, 0.f, 0.f};
    v8f acc[2][4];
#pragma unroll
    for (int i = 0; i < 2; ++i)
#pragma unroll
        for (int j = 0; j < 4; ++j) acc[i][j] = zero;

    for (int kB = 0; kB < K; kB += 32) {
        // A fragments: 16x32 bf16; lane 0-15 = rows, VGPR v holds K pair
        // k(v) = 2v + 16*laneHalf   (contiguous, matches B)
        FragB16 a[2];
#pragma unroll
        for (int i = 0; i < 2; ++i) {
            const unsigned short* ap = Ae + (long long)(rowT + i * 16 + r) * lda + kB;
#pragma unroll
            for (int v = 0; v < 8; ++v) {
                int k0 = 2 * v + h * 16;
                a[i].u[v] = *reinterpret_cast<const unsigned int*>(ap + k0);
            }
        }
        // B fragments: 32x16 bf16; lane 0-15 = cols with K=0..15, lanes 16-31 K=16..31
        // WT is N x K row-major -> 16 consecutive bf16 per lane
        FragB16 b[4];
#pragma unroll
        for (int j = 0; j < 4; ++j) {
            const unsigned short* bp = We + (long long)(colT + j * 16 + r) * K + kB + h * 16;
#pragma unroll
            for (int v = 0; v < 8; ++v)
                b[j].u[v] = *reinterpret_cast<const unsigned int*>(bp + 2 * v);
        }
#pragma unroll
        for (int i = 0; i < 2; ++i)
#pragma unroll
            for (int j = 0; j < 4; ++j)
                acc[i][j] = __builtin_amdgcn_wmma_f32_16x16x32_bf16(
                    false, a[i].v, false, b[j].v, (short)0, acc[i][j], false, false);
    }

    // epilogue: C frag VGPR rr -> row rr + 8*laneHalf, col = lane&15
    const float* x0e = x0f ? x0f + (long long)e * xStride : nullptr;
    const float* xle = xlf ? xlf + (long long)e * xStride : nullptr;
    float*          oF = outF ? outF + (long long)e * outStride : nullptr;
    unsigned short* oH = outH ? outH + (long long)e * outStride : nullptr;
#pragma unroll
    for (int i = 0; i < 2; ++i) {
#pragma unroll
        for (int j = 0; j < 4; ++j) {
#pragma unroll
            for (int rr = 0; rr < 8; ++rr) {
                int m = rowT + i * 16 + rr + h * 8;
                int n = colT + j * 16 + r;
                long long idx = (long long)m * N + n;
                float cv = acc[i][j][rr] + be[n];
                if (mode == 0) {
                    float v = x0e[idx] * cv + xle[idx];
                    oF[idx] = v;
                    oH[idx] = f32_to_bf16u(v);
                } else if (mode == 1) {
                    oF[idx] = cv > 0.f ? cv : 0.f;
                } else {
                    oH[idx] = f32_to_bf16u(cv > 0.f ? cv : 0.f);
                }
            }
        }
    }
}

// ---------------- branch mean (relu already applied) ----------------
__global__ void mean_branches(const float* __restrict__ hE, unsigned short* __restrict__ hm) {
    long long gid = (long long)blockIdx.x * blockDim.x + threadIdx.x;
    const long long per = (long long)BB * MM1;
    if (gid >= per) return;
    float v = (hE[gid] + hE[gid + per] + hE[gid + 2 * per]) * (1.f / 3.f);
    hm[gid] = f32_to_bf16u(v);
}

// ---------------- final head: dot-128 + bias + sigmoid ----------------
__global__ void head_kernel(const unsigned short* __restrict__ h3,
                            const float* __restrict__ W3, const float* __restrict__ b3,
                            float* __restrict__ out) {
    int b = blockIdx.x * blockDim.x + threadIdx.x;
    if (b >= BB) return;
    float p = b3[0];
#pragma unroll 4
    for (int k = 0; k < MM3; ++k)
        p += bf16u_to_f32(h3[(long long)b * MM3 + k]) * W3[k];
    out[b] = 1.f / (1.f + __expf(-p));
}

// ---------------- gram: m[pair,k] = sum_b es_i[b,k,:]^T es_j[b,k,:] ----------------
__global__ __launch_bounds__(256)
void gram_kernel(const float* __restrict__ esF, float* __restrict__ gram) {
    __shared__ float As[64][DD];
    __shared__ float Cs[64][DD];
    const int pk = blockIdx.x;      // pair*16 + feature
    const int pair = pk >> 4;
    const int k = pk & 15;
    const int PI[3] = {1, 2, 2};
    const int PJ[3] = {0, 0, 1};
    const float* ei = esF + (long long)PI[pair] * BB * EOD;
    const float* ej = esF + (long long)PJ[pair] * BB * EOD;
    const int tid = threadIdx.x;
    const int dd = tid >> 3;        // 0..31
    const int e0 = (tid & 7) * 4;   // 0..28
    float acc[4] = {0.f, 0.f, 0.f, 0.f};
    for (int b0 = 0; b0 < BB; b0 += 64) {
        for (int t = tid; t < 64 * DD; t += 256) {
            int bb = t >> 5, d = t & 31;
            long long o = (long long)(b0 + bb) * EOD + k * DD + d;
            As[bb][d] = ei[o];
            Cs[bb][d] = ej[o];
        }
        __syncthreads();
        for (int bb = 0; bb < 64; ++bb) {
            float av = As[bb][dd];
#pragma unroll
            for (int c = 0; c < 4; ++c) acc[c] += av * Cs[bb][e0 + c];
        }
        __syncthreads();
    }
    float* g = gram + (long long)pk * (DD * DD);
#pragma unroll
    for (int c = 0; c < 4; ++c) g[dd * DD + e0 + c] = acc[c];
}

// ---------------- singular values via Jacobi on G = m^T m (one wave / matrix) ----------------
__global__ __launch_bounds__(32)
void svd_jacobi(const float* __restrict__ gram, float* __restrict__ sim) {
    __shared__ float Mm[DD][DD];
    __shared__ float G[DD][DD + 1];
    const int pk = blockIdx.x;
    const int pair = pk >> 4;
    const int t = threadIdx.x;
    const float* m = gram + (long long)pk * (DD * DD);
    for (int i = t; i < DD * DD; i += 32) Mm[i >> 5][i & 31] = m[i];
    __syncthreads();
    for (int p = 0; p < DD; ++p) {
        float s = 0.f;
        for (int d = 0; d < DD; ++d) s += Mm[d][p] * Mm[d][t];
        G[p][t] = s;
    }
    __syncthreads();
    for (int sweep = 0; sweep < 12; ++sweep) {
        for (int p = 0; p < DD - 1; ++p) {
            for (int q = p + 1; q < DD; ++q) {
                float app = G[p][p], aqq = G[q][q], apq = G[p][q];
                float c, s;
                if (__builtin_fabsf(apq) > 1e-20f) {
                    float tau = (aqq - app) / (2.f * apq);
                    float tt = (tau >= 0.f ? 1.f : -1.f) /
                               (__builtin_fabsf(tau) + __builtin_sqrtf(1.f + tau * tau));
                    c = __frsqrt_rn(1.f + tt * tt);
                    s = tt * c;
                } else { c = 1.f; s = 0.f; }
                __syncthreads();
                float gpj = G[p][t], gqj = G[q][t];
                __syncthreads();
                G[p][t] = c * gpj - s * gqj;
                G[q][t] = s * gpj + c * gqj;
                __syncthreads();
                float gjp = G[t][p], gjq = G[t][q];
                __syncthreads();
                G[t][p] = c * gjp - s * gjq;
                G[t][q] = s * gjp + c * gjq;
                __syncthreads();
            }
        }
    }
    float sv = __builtin_sqrtf(fmaxf(G[t][t], 0.f));
    atomicAdd(&sim[pair * DD + t], sv * (1.f / 16.f));   // mean over 16 features
}

__global__ void zero_sim(float* __restrict__ sim) {
    if (threadIdx.x < 96) sim[threadIdx.x] = 0.f;
}

__global__ void reg_reduce(const float* __restrict__ sim, float* __restrict__ out) {
    __shared__ float s[128];
    int t = threadIdx.x;
    float v = (t < 96) ? sim[t] * sim[t] : 0.f;
    s[t] = v;
    __syncthreads();
    for (int st = 64; st > 0; st >>= 1) {
        if (t < st) s[t] += s[t + st];
        __syncthreads();
    }
    if (t == 0) out[0] = REGW * (s[0] / 96.f);
}

// ---------------- launcher ----------------
extern "C" void kernel_launch(void* const* d_in, const int* in_sizes, int n_in,
                              void* d_out, int out_size, void* d_ws, size_t ws_size,
                              hipStream_t stream) {
    const int*   x      = (const int*)d_in[0];
    const float* emb    = (const float*)d_in[1];
    const float* crossW = (const float*)d_in[2];
    const float* crossB = (const float*)d_in[3];
    const float* m1W    = (const float*)d_in[4];
    const float* m1b    = (const float*)d_in[5];
    const float* W1     = (const float*)d_in[6];
    const float* b1     = (const float*)d_in[7];
    const float* W2     = (const float*)d_in[8];
    const float* b2     = (const float*)d_in[9];
    const float* W3     = (const float*)d_in[10];
    const float* b3     = (const float*)d_in[11];
    float* out = (float*)d_out;

    char* ws = (char*)d_ws;
    size_t off = 0;
    auto alloc = [&](size_t bytes) -> void* {
        size_t o = (off + 255) & ~(size_t)255;
        off = o + bytes;
        return (void*)(ws + o);
    };
    const long long sBE = (long long)BB * EOD;               // per-branch activation stride
    float*          esF = (float*)alloc((size_t)NEB * sBE * 4);
    unsigned short* esH = (unsigned short*)alloc((size_t)NEB * sBE * 2);
    float*          xaF = (float*)alloc((size_t)NEB * sBE * 4);
    unsigned short* xaH = (unsigned short*)alloc((size_t)NEB * sBE * 2);
    float*          xbF = (float*)alloc((size_t)NEB * sBE * 4);
    unsigned short* xbH = (unsigned short*)alloc((size_t)NEB * sBE * 2);
    unsigned short* wtC  = (unsigned short*)alloc((size_t)NEB * NLY * EOD * EOD * 2);
    unsigned short* wtM1 = (unsigned short*)alloc((size_t)NEB * EOD * MM1 * 2);
    unsigned short* wt1  = (unsigned short*)alloc((size_t)MM2 * MM1 * 2);
    unsigned short* wt2  = (unsigned short*)alloc((size_t)MM3 * MM2 * 2);
    float*          gram = (float*)alloc((size_t)48 * DD * DD * 4);
    float*          sim  = (float*)alloc(96 * 4);
    // reuse: after cross layer 2, xb* buffers are dead
    float*          hE = xbF;                                 // (NE, B, M1) fp32
    unsigned short* hm = xbH;                                 // (B, M1) bf16
    unsigned short* h2 = xbH + (size_t)BB * MM1;              // (B, M2) bf16
    unsigned short* h3 = h2 + (size_t)BB * MM2;               // (B, M3) bf16

    // 1) weight prep + gather + init
    transpose_to_bf16<<<dim3((EOD * EOD + 255) / 256, 1, NEB * NLY), 256, 0, stream>>>(crossW, wtC, EOD, EOD);
    transpose_to_bf16<<<dim3((EOD * MM1 + 255) / 256, 1, NEB), 256, 0, stream>>>(m1W, wtM1, EOD, MM1);
    transpose_to_bf16<<<dim3((MM1 * MM2 + 255) / 256, 1, 1), 256, 0, stream>>>(W1, wt1, MM1, MM2);
    transpose_to_bf16<<<dim3((MM2 * MM3 + 255) / 256, 1, 1), 256, 0, stream>>>(W2, wt2, MM2, MM3);
    gather_emb<<<(NEB * BB * EOD + 255) / 256, 256, 0, stream>>>(x, emb, esF, esH);
    zero_sim<<<1, 96, 0, stream>>>(sim);

    // 2) DCN cross layers: xl = x0*(xl@W + b) + xl  (ping-pong es -> xa -> xb -> xa)
    dim3 gC(BB / 128, EOD / 128, NEB);
    gemm_bf16_wmma<<<gC, 256, 0, stream>>>(esH, sBE, EOD, wtC + 0 * (size_t)EOD * EOD,
        (long long)NLY * EOD * EOD, crossB + 0 * EOD, (long long)NLY * EOD,
        esF, esF, sBE, xaF, xaH, sBE, EOD, EOD, 0);
    gemm_bf16_wmma<<<gC, 256, 0, stream>>>(xaH, sBE, EOD, wtC + 1 * (size_t)EOD * EOD,
        (long long)NLY * EOD * EOD, crossB + 1 * EOD, (long long)NLY * EOD,
        esF, xaF, sBE, xbF, xbH, sBE, EOD, EOD, 0);
    gemm_bf16_wmma<<<gC, 256, 0, stream>>>(xbH, sBE, EOD, wtC + 2 * (size_t)EOD * EOD,
        (long long)NLY * EOD * EOD, crossB + 2 * EOD, (long long)NLY * EOD,
        esF, xbF, sBE, xaF, xaH, sBE, EOD, EOD, 0);

    // 3) per-branch mlp1 + relu (writes hE, overwriting dead xbF)
    gemm_bf16_wmma<<<dim3(BB / 128, MM1 / 128, NEB), 256, 0, stream>>>(xaH, sBE, EOD,
        wtM1, (long long)EOD * MM1, m1b, MM1,
        nullptr, nullptr, 0, hE, nullptr, (long long)BB * MM1, MM1, EOD, 1);

    // 4) branch mean -> bf16, then MLP tower
    mean_branches<<<(BB * MM1 + 255) / 256, 256, 0, stream>>>(hE, hm);
    gemm_bf16_wmma<<<dim3(BB / 128, MM2 / 128, 1), 256, 0, stream>>>(hm, 0, MM1,
        wt1, 0, b1, 0, nullptr, nullptr, 0, nullptr, h2, 0, MM2, MM1, 2);
    gemm_bf16_wmma<<<dim3(BB / 128, MM3 / 128, 1), 256, 0, stream>>>(h2, 0, MM2,
        wt2, 0, b2, 0, nullptr, nullptr, 0, nullptr, h3, 0, MM3, MM2, 2);
    head_kernel<<<BB / 256, 256, 0, stream>>>(h3, W3, b3, out);

    // 5) regularizer: gram -> Jacobi singular values -> mean-square
    gram_kernel<<<48, 256, 0, stream>>>(esF, gram);
    svd_jacobi<<<48, 32, 0, stream>>>(gram, sim);
    reg_reduce<<<1, 128, 0, stream>>>(sim, out + BB);
}
